// TrajPreLocalAttnLong_56573309223051
// MI455X (gfx1250) — matrix operand order repaired and checked by reading, not compiled, not run on hardware
//
#include <hip/hip_runtime.h>
#include <hip/hip_bf16.h>

typedef __attribute__((ext_vector_type(16))) _Float16 v16h;
typedef __attribute__((ext_vector_type(8)))  _Float16 v8h;
typedef __attribute__((ext_vector_type(8)))  float    v8f;
typedef __attribute__((ext_vector_type(4)))  unsigned v4u;

#define BB   1024   // batch
#define TT   256    // max time steps
#define EE   100    // emb dim
#define HH   100    // hidden
#define DD   12     // time/space slot emb dim
#define FH   400    // 4*H
#define EP   128    // E padded to K-multiple of 32
#define DP   32     // D padded
#define PSTR 624    // preact row stride (halves): 400 gates + 112 tpre + 112 spre
#define TOFF 400
#define SOFF 512

// Pre-swizzled f16 weight tiles in workspace (units: halves).
// Each 16x16x32 B-fragment tile = 512 halves, laid out so lane l reads
// 16 contiguous halves at tile + l*16.
#define WIH_OFF 0        // 25 ntiles * 4 kchunks = 100 tiles
#define WXT_OFF 51200    // 7 * 4 = 28 tiles
#define WTT_OFF 65536    // 7 * 1 = 7 tiles
#define WXS_OFF 69120    // 28 tiles
#define WSS_OFF 83456    // 7 tiles
#define WHH_OFF 87040    // 100 tiles
#define WS_WEIGHT_HALVES 138240

__device__ __forceinline__ float sigmoidf(float x) {
  return 1.0f / (1.0f + __expf(-x));
}

// A-fragment (16x32 f16, M x K) from row-major LDS tile [16][stride].
// Layout per ISA: lanes 0-15 -> M=lane, elems 0..7 = K(32kc+0..7), 8..15 = K(32kc+16..23)
//                 lanes16-31 -> M=lane-16, elems 0..7 = K(32kc+8..15), 8..15 = K(32kc+24..31)
__device__ __forceinline__ v16h loadA(const _Float16* s, int stride, int kc, int half, int lm) {
  union { v16h v; v8h h[2]; } u;
  const _Float16* p = s + lm * stride + kc * 32 + half * 8;
  u.h[0] = *(const v8h*)(p);
  u.h[1] = *(const v8h*)(p + 16);
  return u.v;
}

// B-fragment from pre-swizzled global tile: lane l loads 16 contiguous halves.
__device__ __forceinline__ v16h loadB(const _Float16* tb, int lane) {
  union { v16h v; v8h h[2]; } u;
  const _Float16* p = tb + lane * 16;
  u.h[0] = *(const v8h*)(p);
  u.h[1] = *(const v8h*)(p + 8);
  return u.v;
}

__device__ __forceinline__ v8f wmma_f16(v16h a, v16h b, v8f c) {
  return __builtin_amdgcn_wmma_f32_16x16x32_f16(false, a, false, b, (short)0, c, false, false);
}

// Issue one 16-byte async global->LDS copy (CDNA5, tracked by ASYNCcnt).
// loff = byte offset within this workgroup's LDS allocation (flat addr truncation),
// goff = per-lane byte offset added to uniform 64-bit base in SGPRs (GVS mode).
__device__ __forceinline__ void async_copy_b128(unsigned loff, unsigned goff, const void* gbase) {
  asm volatile("global_load_async_to_lds_b128 %0, %1, %2"
               :: "v"(loff), "v"(goff), "s"(gbase) : "memory");
}
__device__ __forceinline__ void wait_async0() {
  asm volatile("s_wait_asynccnt 0x0" ::: "memory");
}

// ---------------------------------------------------------------------------
// Kernel 1: convert + swizzle all weight matrices into B-fragment layout.
// grid = 270 tiles, block = 256.
// ---------------------------------------------------------------------------
__global__ void __launch_bounds__(256) swizzle_weights(
    const float* __restrict__ Wih, const float* __restrict__ Whh,
    const float* __restrict__ Wxt, const float* __restrict__ Wtt,
    const float* __restrict__ Wxs, const float* __restrict__ Wss,
    _Float16* __restrict__ wsw)
{
  int tid = blockIdx.x;
  const float* src; int off, kcPer, kmax, nmax, stride, lt;
  if (tid < 100)      { lt = tid;       src = Wih; off = WIH_OFF; kcPer = 4; kmax = EE; nmax = FH; stride = FH; }
  else if (tid < 128) { lt = tid - 100; src = Wxt; off = WXT_OFF; kcPer = 4; kmax = EE; nmax = HH; stride = HH; }
  else if (tid < 135) { lt = tid - 128; src = Wtt; off = WTT_OFF; kcPer = 1; kmax = DD; nmax = HH; stride = HH; }
  else if (tid < 163) { lt = tid - 135; src = Wxs; off = WXS_OFF; kcPer = 4; kmax = EE; nmax = HH; stride = HH; }
  else if (tid < 170) { lt = tid - 163; src = Wss; off = WSS_OFF; kcPer = 1; kmax = DD; nmax = HH; stride = HH; }
  else                { lt = tid - 170; src = Whh; off = WHH_OFF; kcPer = 4; kmax = HH; nmax = FH; stride = FH; }
  int ntile = lt / kcPer, kc = lt % kcPer;
  _Float16* dst = wsw + off + (size_t)lt * 512;
  for (int idx = threadIdx.x; idx < 512; idx += 256) {
    int l = idx >> 4, e = idx & 15;
    int k = kc * 32 + ((l >> 4) << 3) + (e & 7) + ((e >> 3) << 4);   // 8*(lane/16) + (e&7) + 16*(e/8)
    int n = ntile * 16 + (l & 15);
    float v = (k < kmax && n < nmax) ? src[(size_t)k * stride + n] : 0.0f;
    dst[idx] = (_Float16)v;
  }
}

// ---------------------------------------------------------------------------
// Kernel 2: precompute all h-independent pre-activations for every (b,t).
// preact[(t*B+b)][0:400]   = x@W_ih + b         (gates i|f|g|o)
// preact[(t*B+b)][400:512] = x@W_xt + Tq@W_tt + b_t (padded 112)
// preact[(t*B+b)][512:624] = x@W_xs + Sq@W_ss + b_s (padded 112)
// grid = B*T/16 blocks; each block: one t, 16 consecutive b rows.
// ---------------------------------------------------------------------------
__global__ void __launch_bounds__(256) precompute_gates(
    const int* __restrict__ traj, const int* __restrict__ trajLenP,
    const int* __restrict__ tu, const int* __restrict__ tl,
    const int* __restrict__ tus, const int* __restrict__ tls,
    const int* __restrict__ su, const int* __restrict__ sl,
    const int* __restrict__ sus, const int* __restrict__ sls,
    const float* __restrict__ embLoc, const float* __restrict__ embT,
    const float* __restrict__ embS,
    const float* __restrict__ bg, const float* __restrict__ bt,
    const float* __restrict__ bs,
    const _Float16* __restrict__ wsw, _Float16* __restrict__ preact)
{
  __shared__ __align__(16) _Float16 xs[16 * EP];     // x tile, f16, K-padded
  __shared__ __align__(16) _Float16 tsb[16 * DP];    // Tq tile
  __shared__ __align__(16) _Float16 ssb[16 * DP];    // Sq tile
  __shared__ __align__(16) _Float16 outb[16 * PSTR]; // staged output

  const int r0 = blockIdx.x * 16;       // flat row = t*B + b
  const int t  = r0 / BB;
  const int b0 = r0 % BB;
  const int steps = trajLenP[0] + 1;
  if (t >= steps) return;
  const int tid = threadIdx.x;

  // gather x = emb_loc[traj[b][t]] -> f16, zero-pad cols [100,128)
  for (int idx = tid; idx < 16 * EP; idx += 256) {
    int row = idx >> 7, col = idx & (EP - 1);
    float v = 0.0f;
    if (col < EE) {
      int loc = traj[(b0 + row) * TT + t];
      v = embLoc[(size_t)loc * EE + col];
    }
    xs[idx] = (_Float16)v;
  }
  // interpolated time/space slot embeddings -> f16, zero-pad cols [12,32)
  for (int idx = tid; idx < 16 * DP; idx += 256) {
    int row = idx >> 5, col = idx & (DP - 1);
    float vt = 0.0f, vs = 0.0f;
    if (col < DD) {
      int base = (b0 + row) * TT + t;
      { float up = (float)tus[base], low = (float)tls[base];
        int iu = tu[base], il = tl[base];
        vt = (up * embT[il * DD + col] + low * embT[iu * DD + col]) / fmaxf(up + low, 1.0f); }
      { float up = (float)sus[base], low = (float)sls[base];
        int iu = su[base], il = sl[base];
        vs = (up * embS[il * DD + col] + low * embS[iu * DD + col]) / fmaxf(up + low, 1.0f); }
    }
    tsb[idx] = (_Float16)vt;
    ssb[idx] = (_Float16)vs;
  }
  __syncthreads();

  const int wave = tid >> 5, lane = tid & 31, half = lane >> 4, lm = lane & 15;
  // 39 N-tiles: 25 gate tiles + 7 tpre tiles + 7 spre tiles, across 8 waves
  for (int tile = wave; tile < 39; tile += 8) {
    v8f c;
    int colbase;
    if (tile < 25) {
      colbase = tile * 16;
      float bv = bg[colbase + lm];
      for (int r = 0; r < 8; ++r) c[r] = bv;
      for (int kc = 0; kc < 4; ++kc)
        c = wmma_f16(loadA(xs, EP, kc, half, lm),
                     loadB(wsw + WIH_OFF + (size_t)(tile * 4 + kc) * 512, lane), c);
    } else if (tile < 32) {
      int nt = tile - 25; colbase = TOFF + nt * 16;
      int ncol = nt * 16 + lm;
      float bv = (ncol < HH) ? bt[ncol] : 0.0f;
      for (int r = 0; r < 8; ++r) c[r] = bv;
      for (int kc = 0; kc < 4; ++kc)
        c = wmma_f16(loadA(xs, EP, kc, half, lm),
                     loadB(wsw + WXT_OFF + (size_t)(nt * 4 + kc) * 512, lane), c);
      c = wmma_f16(loadA(tsb, DP, 0, half, lm),
                   loadB(wsw + WTT_OFF + (size_t)nt * 512, lane), c);
    } else {
      int nt = tile - 32; colbase = SOFF + nt * 16;
      int ncol = nt * 16 + lm;
      float bv = (ncol < HH) ? bs[ncol] : 0.0f;
      for (int r = 0; r < 8; ++r) c[r] = bv;
      for (int kc = 0; kc < 4; ++kc)
        c = wmma_f16(loadA(xs, EP, kc, half, lm),
                     loadB(wsw + WXS_OFF + (size_t)(nt * 4 + kc) * 512, lane), c);
      c = wmma_f16(loadA(ssb, DP, 0, half, lm),
                   loadB(wsw + WSS_OFF + (size_t)nt * 512, lane), c);
    }
    for (int r = 0; r < 8; ++r)
      outb[(r + 8 * half) * PSTR + colbase + lm] = (_Float16)c[r];
  }
  __syncthreads();

  // coalesced non-temporal store: 16 rows * 624 halves contiguous in global
  // (327 MB stream > 192 MB L2 -> don't pollute the cache)
  const v4u* srcv = (const v4u*)outb;
  v4u* dstv = (v4u*)(preact + (size_t)r0 * PSTR);
  for (int i = tid; i < (16 * PSTR) / 8; i += 256)
    __builtin_nontemporal_store(srcv[i], &dstv[i]);
}

// ---------------------------------------------------------------------------
// Kernel 3: sequential LSTM recurrence. 64 blocks; block owns 16 batch rows
// for all steps.
//  - W_hh B-fragments hoisted into registers (<=16 v16h per thread), loaded once.
//  - preact slabs double-buffered in LDS via global_load_async_to_lds_b128,
//    prefetching step t+1 while computing step t (ASYNCcnt + barrier sync).
//  - cell state c in registers; gate exchange buffer in f16 LDS.
// ---------------------------------------------------------------------------
__global__ void __launch_bounds__(256) lstm_recurrence(
    const int* __restrict__ trajLenP, const _Float16* __restrict__ wsw,
    const _Float16* __restrict__ preact, float* __restrict__ out)
{
  __shared__ __align__(16) _Float16 hbuf[16 * EP];        // h, f16, K-padded (4 KB)
  __shared__ __align__(16) _Float16 sumb[16 * FH];        // gate preacts (12.8 KB)
  __shared__ __align__(16) _Float16 pslab[2][16 * PSTR];  // preact slabs (2x20 KB)

  const int tid = threadIdx.x;
  const int b0 = blockIdx.x * 16;
  const int steps = trajLenP[0] + 1;
  const int wave = tid >> 5, lane = tid & 31, half = lane >> 4, lm = lane & 15;
  const int NCHUNK = (16 * PSTR) / 8;   // 16-byte chunks per slab = 1248

  // Hoist W_hh B-fragments into registers: wave w owns tiles {w, w+8, w+16[, w+24]}.
  const _Float16* whh = wsw + WHH_OFF;
  v16h breg[4][4];
#pragma unroll
  for (int i = 0; i < 4; ++i) {
    int tile = wave + 8 * i;
    if (tile < 25) {
#pragma unroll
      for (int kc = 0; kc < 4; ++kc)
        breg[i][kc] = loadB(whh + (size_t)(tile * 4 + kc) * 512, lane);
    }
  }

  // cell state in registers: element idx = tid + 256*k
  float creg[7];
#pragma unroll
  for (int k = 0; k < 7; ++k) creg[k] = 0.0f;

  for (int idx = tid; idx < 16 * EP; idx += 256) hbuf[idx] = (_Float16)0.0f;

  // prefetch step-0 slab into buffer 0
  {
    const void* g0 = (const void*)(preact + (size_t)b0 * PSTR);
    unsigned lbase = (unsigned)(uintptr_t)(&pslab[0][0]);
    for (int k = 0; k < 5; ++k) {
      int chunk = tid + 256 * k;
      if (chunk < NCHUNK)
        async_copy_b128(lbase + chunk * 16u, chunk * 16u, g0);
    }
  }
  wait_async0();
  __syncthreads();

  for (int t = 0; t < steps; ++t) {
    const int cur = t & 1;
    const _Float16* ps = &pslab[cur][0];

    // async prefetch of step t+1 into the other buffer (readers of that
    // buffer finished at the barrier that ended step t-1)
    if (t + 1 < steps) {
      const void* gn = (const void*)(preact + ((size_t)(t + 1) * BB + b0) * PSTR);
      unsigned lbase = (unsigned)(uintptr_t)(&pslab[cur ^ 1][0]);
      for (int k = 0; k < 5; ++k) {
        int chunk = tid + 256 * k;
        if (chunk < NCHUNK)
          async_copy_b128(lbase + chunk * 16u, chunk * 16u, gn);
      }
    }

    // A fragments of h_{t-1} (shared by all of this wave's tiles)
    v16h areg[4];
#pragma unroll
    for (int kc = 0; kc < 4; ++kc) areg[kc] = loadA(hbuf, EP, kc, half, lm);

    // gates = preact + h @ W_hh   (25 N-tiles of 16, K = 128 padded)
#pragma unroll
    for (int i = 0; i < 4; ++i) {
      int tile = wave + 8 * i;
      if (tile < 25) {
        int colbase = tile * 16;
        v8f c;
#pragma unroll
        for (int r = 0; r < 8; ++r)
          c[r] = (float)ps[(r + 8 * half) * PSTR + colbase + lm];
#pragma unroll
        for (int kc = 0; kc < 4; ++kc)
          c = wmma_f16(areg[kc], breg[i][kc], c);
#pragma unroll
        for (int r = 0; r < 8; ++r)
          sumb[(r + 8 * half) * FH + colbase + lm] = (_Float16)c[r];
      }
    }
    __syncthreads();

    // elementwise LSTM cell update
#pragma unroll
    for (int k = 0; k < 7; ++k) {
      int idx = tid + 256 * k;
      if (idx < 16 * HH) {
        int row = idx / HH, j = idx - row * HH;
        const _Float16* sr = sumb + row * FH;
        float iv = sigmoidf((float)sr[j]);
        float fv = sigmoidf((float)sr[HH + j]);
        float gv = tanhf((float)sr[2 * HH + j]);
        float ov = sigmoidf((float)sr[3 * HH + j]);
        const _Float16* pr = ps + row * PSTR;
        float tg = sigmoidf((float)pr[TOFF + j]);
        float sg = sigmoidf((float)pr[SOFF + j]);
        float cv = fv * creg[k] + iv * tg * sg * gv;
        creg[k] = cv;
        float hv = ov * tanhf(cv);
        hbuf[row * EP + j] = (_Float16)hv;
        if (t == steps - 1) out[(b0 + row) * HH + j] = hv;
      }
    }
    // next-step slab must be fully in LDS before anyone reads it next iter
    wait_async0();
    __syncthreads();
  }
}

// ---------------------------------------------------------------------------
extern "C" void kernel_launch(void* const* d_in, const int* in_sizes, int n_in,
                              void* d_out, int out_size, void* d_ws, size_t ws_size,
                              hipStream_t stream) {
  (void)in_sizes; (void)n_in; (void)out_size; (void)ws_size;
  const int*   traj    = (const int*)d_in[0];
  // d_in[1] = lennew (unused by reference)
  const int*   trajLen = (const int*)d_in[2];
  const int*   tu  = (const int*)d_in[3];
  const int*   tl  = (const int*)d_in[4];
  const int*   tus = (const int*)d_in[5];
  const int*   tls = (const int*)d_in[6];
  const int*   su  = (const int*)d_in[7];
  const int*   sl  = (const int*)d_in[8];
  const int*   sus = (const int*)d_in[9];
  const int*   sls = (const int*)d_in[10];
  const float* embLoc = (const float*)d_in[11];
  const float* embT   = (const float*)d_in[12];
  const float* embS   = (const float*)d_in[13];
  const float* Wih = (const float*)d_in[14];
  const float* Whh = (const float*)d_in[15];
  const float* bg  = (const float*)d_in[16];
  const float* Wxt = (const float*)d_in[17];
  const float* Wtt = (const float*)d_in[18];
  const float* bt  = (const float*)d_in[19];
  const float* Wxs = (const float*)d_in[20];
  const float* Wss = (const float*)d_in[21];
  const float* bs  = (const float*)d_in[22];

  _Float16* wsw    = (_Float16*)d_ws;
  _Float16* preact = wsw + WS_WEIGHT_HALVES;   // 327 MB: [T][B][624] f16

  swizzle_weights<<<270, 256, 0, stream>>>(Wih, Whh, Wxt, Wtt, Wxs, Wss, wsw);

  precompute_gates<<<(BB * TT) / 16, 256, 0, stream>>>(
      traj, trajLen, tu, tl, tus, tls, su, sl, sus, sls,
      embLoc, embT, embS, bg, bt, bs, wsw, preact);

  lstm_recurrence<<<BB / 16, 256, 0, stream>>>(trajLen, wsw, preact, (float*)d_out);
}
